// FRUnrolled_36455682408728
// MI455X (gfx1250) — compile-verified, matrix operand order checked by reading, not run on hardware
//
#include <hip/hip_runtime.h>

// ---------------------------------------------------------------------------
// FR-unrolled GNN layout pipeline for gfx1250 (MI455X).
// Edge MLPs fused (gather -> GEMM -> relu -> GEMM -> relu -> scatter-add)
// using v_wmma_f32_16x16x32_bf16. Weights staged into LDS via the Tensor Data
// Mover (tensor_load_to_lds + s_wait_tensorcnt) when available, cooperative
// vector copy otherwise. Relu layout exchange through per-wave LDS tiles.
// All-pairs repulsion is LDS-resident per group.
// ---------------------------------------------------------------------------

#define G_GRP   64
#define NG_GRP  512
#define N_NODES (G_GRP * NG_GRP)   // 32768
#define E_TOT   262144
#define UND_E   (E_TOT / 2)        // 131072
#define EPS_F   0.01f
#define CLAMP_F 0.05f

typedef __attribute__((ext_vector_type(16))) __bf16 v16bf;
typedef __attribute__((ext_vector_type(8)))  __bf16 v8bf;
typedef __attribute__((ext_vector_type(8)))  float  v8f;

// ---- Tensor Data Mover staging ---------------------------------------------
#if defined(__AMDGCN__) && __has_builtin(__builtin_amdgcn_tensor_load_to_lds) && \
    __has_builtin(__builtin_amdgcn_s_wait_tensorcnt)
#define HAVE_TDM 1
#endif

#if defined(HAVE_TDM)
typedef __attribute__((ext_vector_type(4))) unsigned int u32x4;
typedef __attribute__((ext_vector_type(8))) int i32x8;
typedef __attribute__((ext_vector_type(4))) int i32x4;

// Issue one TDM 2-D tile load (contiguous rows) of rowBytes x rows from
// global memory into LDS. Descriptor layout per CDNA5 ISA 08_async_tensor §8.
// All fields are wave-uniform -> SGPR groups as required by the ISA.
__device__ __forceinline__ void tdm_load_2d(void* lds_dst, const void* gsrc,
                                            unsigned rowBytes, unsigned rows) {
  const unsigned lds = (unsigned)(unsigned long long)lds_dst;   // LDS offset
  const unsigned long long ga = (unsigned long long)gsrc;
  u32x4 g0;
  g0[0] = 1u;                                        // count=1 (valid), user mode
  g0[1] = lds;                                       // lds_addr
  g0[2] = (unsigned)(ga & 0xFFFFFFFFu);              // global_addr[31:0]
  g0[3] = (unsigned)((ga >> 32) & 0x1FFFFFFu)        // global_addr[56:32]
          | (2u << 30);                              // type=2 ("image")
  i32x8 g1;
  g1[0] = 0;                                         // wg_mask=0, data_size=0 (1B)
  g1[1] = (int)((rowBytes & 0xFFFFu) << 16);         // tensor_dim0[15:0] @63:48
  g1[2] = (int)((rowBytes >> 16) |                   // tensor_dim0[31:16]
                ((rows & 0xFFFFu) << 16));           // tensor_dim1[15:0]
  g1[3] = (int)((rows >> 16) |                       // tensor_dim1[31:16]
                ((rowBytes & 0xFFFFu) << 16));       // tile_dim0
  g1[4] = (int)(rows & 0xFFFFu);                     // tile_dim1 (tile_dim2=0)
  g1[5] = (int)rowBytes;                             // tensor_dim0_stride[31:0]
  g1[6] = 0;                                         // stride hi, dim1_stride lo
  g1[7] = 0;
  const i32x4 gz = {0, 0, 0, 0};
#if __clang_major__ >= 23
  const i32x8 gz8 = {0, 0, 0, 0, 0, 0, 0, 0};
  __builtin_amdgcn_tensor_load_to_lds(g0, g1, gz, gz, gz8, 0);
#else
  __builtin_amdgcn_tensor_load_to_lds(g0, g1, gz, gz, 0);
#endif
}
#define TDM_WAIT() __builtin_amdgcn_s_wait_tensorcnt(0)
#endif

// ---- WMMA helpers ----------------------------------------------------------
// A fragment (16x32 bf16) lane layout (ISA 7.12.2):
//   lane&15 = M row; halves[0..7]  = K k0+sub+0..7
//                    halves[8..15] = K k0+16+sub+0..7, sub = (lane>>4)*8
// B fragment for B^T stored row-major (Nout x K): same pattern with row = n.
__device__ __forceinline__ v16bf frag_ld(const __bf16* rowbase, int k0, int sub) {
  union U { v16bf v; v8bf h[2]; } u;
  u.h[0] = *(const v8bf*)(rowbase + k0 + sub);
  u.h[1] = *(const v8bf*)(rowbase + k0 + sub + 16);
  return u.v;
}

__device__ __forceinline__ v8f wmma_bf16(v16bf a, v16bf b, v8f c) {
  return __builtin_amdgcn_wmma_f32_16x16x32_bf16(false, a, false, b, (short)0,
                                                 c, false, false);
}

// ---- small utility kernels -------------------------------------------------
__global__ void k_zero_f32(float* p, int n) {
  int i = blockIdx.x * blockDim.x + threadIdx.x;
  if (i < n) p[i] = 0.0f;
}
__global__ void k_zero_bf(__bf16* p, int n) {
  int i = blockIdx.x * blockDim.x + threadIdx.x;
  if (i < n) p[i] = (__bf16)0.0f;
}
// w: (K x Nout) f32 row-major  ->  out: (Nout x Kpad) bf16 row-major (transposed)
__global__ void k_pack_wt(const float* __restrict__ w, __bf16* __restrict__ out,
                          int Nout, int Kpad, int dstK0, int srcK0, int copyK) {
  int i = blockIdx.x * blockDim.x + threadIdx.x;
  if (i >= Nout * copyK) return;
  int n = i / copyK, k = i % copyK;
  out[(size_t)n * Kpad + dstK0 + k] = (__bf16)w[(size_t)(srcK0 + k) * Nout + n];
}
// h = concat(x[:, -2:], x[:, :-2]) padded to 32 cols (bf16) + f32 coords
__global__ void k_pack_h(const float* __restrict__ x, __bf16* __restrict__ Hpad,
                         float* __restrict__ coords) {
  int i = blockIdx.x * blockDim.x + threadIdx.x;
  if (i >= N_NODES) return;
  const float* xi = x + (size_t)i * 18;
  __bf16* hp = Hpad + (size_t)i * 32;
  hp[0] = (__bf16)xi[16];
  hp[1] = (__bf16)xi[17];
#pragma unroll
  for (int j = 0; j < 16; ++j) hp[2 + j] = (__bf16)xi[j];
#pragma unroll
  for (int j = 18; j < 32; ++j) hp[j] = (__bf16)0.0f;
  coords[i * 2 + 0] = xi[16];
  coords[i * 2 + 1] = xi[17];
}
__global__ void k_deg(const int* __restrict__ row, float* __restrict__ deg) {
  int e = blockIdx.x * blockDim.x + threadIdx.x;
  if (e < E_TOT) atomicAdd(&deg[row[e]], 1.0f);
}
__global__ void k_counts(const int* __restrict__ batch, float* __restrict__ counts) {
  int i = blockIdx.x * blockDim.x + threadIdx.x;
  if (i < N_NODES) atomicAdd(&counts[batch[i]], 1.0f);
}
__global__ void k_kg(const float* __restrict__ counts, float* __restrict__ kg) {
  int g = blockIdx.x * blockDim.x + threadIdx.x;
  if (g < G_GRP) kg[g] = rsqrtf(fmaxf(counts[g], 1.0f));
}
// agg / deg_c -> bf16
__global__ void k_finish_agg(const float* __restrict__ agg,
                             const float* __restrict__ deg,
                             __bf16* __restrict__ aggb) {
  int i = blockIdx.x * blockDim.x + threadIdx.x;
  if (i >= N_NODES * 64) return;
  int node = i >> 6;
  aggb[i] = (__bf16)(agg[i] / fmaxf(deg[node], 1.0f));
}
__global__ void k_copy_out(const float* __restrict__ coords, float* __restrict__ out) {
  int i = blockIdx.x * blockDim.x + threadIdx.x;
  if (i < N_NODES * 2) out[i] = coords[i];
}

// ---- fused edge MLP: agg[row] += relu(W2 relu(W1 Zin[col] + b1) + b2) ------
// K1 in {32, 64}; output width 64. One 16-edge tile per wave.
__global__ void __launch_bounds__(128)
k_edge_mlp(const __bf16* __restrict__ Zin, int K1,
           const int* __restrict__ gatherIdx, const int* __restrict__ scatterIdx,
           const __bf16* __restrict__ W1t, const float* __restrict__ b1,
           const __bf16* __restrict__ W2t, const float* __restrict__ b2,
           float* __restrict__ agg) {
  __shared__ __attribute__((aligned(16))) __bf16 sW1[64 * 64];
  __shared__ __attribute__((aligned(16))) __bf16 sW2[64 * 64];
  __shared__ __attribute__((aligned(16))) __bf16 sT[4][16 * 64];
  const int tid = threadIdx.x;
#if defined(HAVE_TDM)
  if (tid == 0) {                       // one wave issues both DMA descriptors
    tdm_load_2d(sW1, W1t, (unsigned)(K1 * 2), 64);
    tdm_load_2d(sW2, W2t, 128, 64);
  }
  TDM_WAIT();
#else
  for (int i = tid; i < 64 * K1 / 8; i += 128) ((v8bf*)sW1)[i] = ((const v8bf*)W1t)[i];
  for (int i = tid; i < 64 * 64 / 8; i += 128) ((v8bf*)sW2)[i] = ((const v8bf*)W2t)[i];
#endif
  __syncthreads();

  const int lane = tid & 31, wv = tid >> 5;
  const int hi = lane >> 4, l15 = lane & 15, sub = hi << 3;
  const int tb = (blockIdx.x * 4 + wv) * 16;
  __builtin_prefetch(gatherIdx + tb + 64, 0, 1);

  const int arow = gatherIdx[tb + l15];
  const __bf16* abase = Zin + (size_t)arow * K1;

  v8f acc[4] = {};
  for (int k0 = 0; k0 < K1; k0 += 32) {
    v16bf a = frag_ld(abase, k0, sub);
#pragma unroll
    for (int t = 0; t < 4; ++t) {
      v16bf b = frag_ld(&sW1[(t * 16 + l15) * K1], k0, sub);
      acc[t] = wmma_bf16(a, b, acc[t]);
    }
  }
#pragma unroll
  for (int t = 0; t < 4; ++t) {
    const float bb = b1[t * 16 + l15];
#pragma unroll
    for (int g = 0; g < 8; ++g)
      sT[wv][(g + hi * 8) * 64 + t * 16 + l15] = (__bf16)fmaxf(acc[t][g] + bb, 0.0f);
  }
  v8f acc2[4] = {};
  const __bf16* tbase = &sT[wv][l15 * 64];
  for (int k0 = 0; k0 < 64; k0 += 32) {
    v16bf a = frag_ld(tbase, k0, sub);
#pragma unroll
    for (int t = 0; t < 4; ++t) {
      v16bf b = frag_ld(&sW2[(t * 16 + l15) * 64], k0, sub);
      acc2[t] = wmma_bf16(a, b, acc2[t]);
    }
  }
#pragma unroll
  for (int g = 0; g < 8; ++g) {
    const int r = scatterIdx[tb + g + hi * 8];
    float* out = agg + (size_t)r * 64;
#pragma unroll
    for (int t = 0; t < 4; ++t)
      atomicAdd(out + t * 16 + l15, fmaxf(acc2[t][g] + b2[t * 16 + l15], 0.0f));
  }
}

// ---- z = Hpad @ W_inp + b --------------------------------------------------
__global__ void __launch_bounds__(128)
k_node_inp(const __bf16* __restrict__ Hpad, const __bf16* __restrict__ Wt,
           const float* __restrict__ bias, float* __restrict__ Zf,
           __bf16* __restrict__ Zb) {
  __shared__ __attribute__((aligned(16))) __bf16 sW[64 * 32];
  const int tid = threadIdx.x;
#if defined(HAVE_TDM)
  if (tid == 0) tdm_load_2d(sW, Wt, 64, 64);
  TDM_WAIT();
#else
  for (int i = tid; i < 64 * 32 / 8; i += 128) ((v8bf*)sW)[i] = ((const v8bf*)Wt)[i];
#endif
  __syncthreads();
  const int lane = tid & 31, wv = tid >> 5;
  const int hi = lane >> 4, l15 = lane & 15, sub = hi << 3;
  const int nb = (blockIdx.x * 4 + wv) * 16;
  v8f acc[4] = {};
  v16bf a = frag_ld(Hpad + (size_t)(nb + l15) * 32, 0, sub);
#pragma unroll
  for (int t = 0; t < 4; ++t) {
    v16bf b = frag_ld(&sW[(t * 16 + l15) * 32], 0, sub);
    acc[t] = wmma_bf16(a, b, acc[t]);
  }
#pragma unroll
  for (int t = 0; t < 4; ++t) {
    const float bb = bias[t * 16 + l15];
#pragma unroll
    for (int g = 0; g < 8; ++g) {
      const float v = acc[t][g] + bb;
      const size_t o = (size_t)(nb + g + hi * 8) * 64 + t * 16 + l15;
      Zf[o] = v;
      Zb[o] = (__bf16)v;
    }
  }
}

// ---- z += W2 relu(W1 [z||agg] + b1) + b2 -----------------------------------
__global__ void __launch_bounds__(128)
k_node_update(float* __restrict__ Zf, __bf16* __restrict__ Zb,
              const __bf16* __restrict__ AggB,
              const __bf16* __restrict__ W1t, const float* __restrict__ b1,
              const __bf16* __restrict__ W2t, const float* __restrict__ b2) {
  __shared__ __attribute__((aligned(16))) __bf16 sW1[64 * 128];
  __shared__ __attribute__((aligned(16))) __bf16 sW2[64 * 64];
  __shared__ __attribute__((aligned(16))) __bf16 sT[4][16 * 64];
  const int tid = threadIdx.x;
#if defined(HAVE_TDM)
  if (tid == 0) {
    tdm_load_2d(sW1, W1t, 256, 64);
    tdm_load_2d(sW2, W2t, 128, 64);
  }
  TDM_WAIT();
#else
  for (int i = tid; i < 64 * 128 / 8; i += 128) ((v8bf*)sW1)[i] = ((const v8bf*)W1t)[i];
  for (int i = tid; i < 64 * 64 / 8; i += 128) ((v8bf*)sW2)[i] = ((const v8bf*)W2t)[i];
#endif
  __syncthreads();
  const int lane = tid & 31, wv = tid >> 5;
  const int hi = lane >> 4, l15 = lane & 15, sub = hi << 3;
  const int nb = (blockIdx.x * 4 + wv) * 16;
  const __bf16* zrow = Zb + (size_t)(nb + l15) * 64;
  const __bf16* grow = AggB + (size_t)(nb + l15) * 64;
  v8f acc[4] = {};
  for (int k0 = 0; k0 < 128; k0 += 32) {
    v16bf a = (k0 < 64) ? frag_ld(zrow, k0, sub) : frag_ld(grow, k0 - 64, sub);
#pragma unroll
    for (int t = 0; t < 4; ++t) {
      v16bf b = frag_ld(&sW1[(t * 16 + l15) * 128], k0, sub);
      acc[t] = wmma_bf16(a, b, acc[t]);
    }
  }
#pragma unroll
  for (int t = 0; t < 4; ++t) {
    const float bb = b1[t * 16 + l15];
#pragma unroll
    for (int g = 0; g < 8; ++g)
      sT[wv][(g + hi * 8) * 64 + t * 16 + l15] = (__bf16)fmaxf(acc[t][g] + bb, 0.0f);
  }
  v8f acc2[4] = {};
  const __bf16* tbase = &sT[wv][l15 * 64];
  for (int k0 = 0; k0 < 64; k0 += 32) {
    v16bf a = frag_ld(tbase, k0, sub);
#pragma unroll
    for (int t = 0; t < 4; ++t) {
      v16bf b = frag_ld(&sW2[(t * 16 + l15) * 64], k0, sub);
      acc2[t] = wmma_bf16(a, b, acc2[t]);
    }
  }
#pragma unroll
  for (int t = 0; t < 4; ++t) {
    const float bb = b2[t * 16 + l15];
#pragma unroll
    for (int g = 0; g < 8; ++g) {
      const size_t o = (size_t)(nb + g + hi * 8) * 64 + t * 16 + l15;
      const float v = Zf[o] + acc2[t][g] + bb;
      Zf[o] = v;
      Zb[o] = (__bf16)v;
    }
  }
}

// ---- edge-weight head: softplus(h2 relu(h1 [z_s||z_d||deg_s||deg_d]+b1)+b2) -
__global__ void __launch_bounds__(128)
k_head(const __bf16* __restrict__ Zb, const int* __restrict__ srcIdx,
       const int* __restrict__ dstIdx, const __bf16* __restrict__ W1t /*32x128*/,
       const float* __restrict__ b1, const float* __restrict__ h1w /*130x32*/,
       const float* __restrict__ h2w, const float* __restrict__ h2b,
       const float* __restrict__ deg, float* __restrict__ ew) {
  __shared__ __attribute__((aligned(16))) __bf16 sW[32 * 128];
  __shared__ __attribute__((aligned(16))) float sH[4][16 * 32];
  const int tid = threadIdx.x;
#if defined(HAVE_TDM)
  if (tid == 0) tdm_load_2d(sW, W1t, 256, 32);
  TDM_WAIT();
#else
  for (int i = tid; i < 32 * 128 / 8; i += 128) ((v8bf*)sW)[i] = ((const v8bf*)W1t)[i];
#endif
  __syncthreads();
  const int lane = tid & 31, wv = tid >> 5;
  const int hi = lane >> 4, l15 = lane & 15, sub = hi << 3;
  const int tb = (blockIdx.x * 4 + wv) * 16;
  const __bf16* srow = Zb + (size_t)srcIdx[tb + l15] * 64;
  const __bf16* drow = Zb + (size_t)dstIdx[tb + l15] * 64;
  v8f acc[2] = {};
  for (int k0 = 0; k0 < 128; k0 += 32) {
    v16bf a = (k0 < 64) ? frag_ld(srow, k0, sub) : frag_ld(drow, k0 - 64, sub);
#pragma unroll
    for (int t = 0; t < 2; ++t) {
      v16bf b = frag_ld(&sW[(t * 16 + l15) * 128], k0, sub);
      acc[t] = wmma_bf16(a, b, acc[t]);
    }
  }
#pragma unroll
  for (int t = 0; t < 2; ++t) {
    const int n = t * 16 + l15;
    const float bb = b1[n], wds = h1w[128 * 32 + n], wdd = h1w[129 * 32 + n];
#pragma unroll
    for (int g = 0; g < 8; ++g) {
      const int m = g + hi * 8;
      const float ds = deg[srcIdx[tb + m]];
      const float dd = deg[dstIdx[tb + m]];
      sH[wv][m * 32 + n] = fmaxf(acc[t][g] + bb + ds * wds + dd * wdd, 0.0f);
    }
  }
  if (lane < 16) {
    float s = h2b[0];
    const float* hr = &sH[wv][lane * 32];
#pragma unroll
    for (int j = 0; j < 32; ++j) s += hr[j] * h2w[j];
    float sp = (s > 20.0f) ? s : log1pf(__expf(s));
    sp += 1e-6f;
    ew[tb + lane] = sp;
    ew[UND_E + tb + lane] = sp;
  }
}

// ---- attractive forces over edges ------------------------------------------
__global__ void k_attract(const float* __restrict__ coords,
                          const int* __restrict__ row, const int* __restrict__ col,
                          const float* __restrict__ ew, const int* __restrict__ batch,
                          const float* __restrict__ kg, float* __restrict__ fnet) {
  const int e = blockIdx.x * blockDim.x + threadIdx.x;
  if (e >= E_TOT) return;
  const int r = row[e], c = col[e];
  const float dx = coords[r * 2 + 0] - coords[c * 2 + 0];
  const float dy = coords[r * 2 + 1] - coords[c * 2 + 1];
  const float dist = sqrtf(dx * dx + dy * dy) + EPS_F;
  const float s = -(dist / kg[batch[r]]) * ew[e];
  const float ax = s * dx, ay = s * dy;
  atomicAdd(&fnet[r * 2 + 0], ax);
  atomicAdd(&fnet[r * 2 + 1], ay);
  atomicAdd(&fnet[c * 2 + 0], -ax);
  atomicAdd(&fnet[c * 2 + 1], -ay);
}

// ---- all-pairs repulsion, one group per block, coords LDS-resident ---------
__global__ void __launch_bounds__(256)
k_repulse(const float* __restrict__ coords, const float* __restrict__ kg,
          float* __restrict__ fnet) {
  __shared__ float sx[NG_GRP], sy[NG_GRP];
  const int g = blockIdx.x, tid = threadIdx.x;
  for (int i = tid; i < NG_GRP; i += 256) {
    sx[i] = coords[(g * NG_GRP + i) * 2 + 0];
    sy[i] = coords[(g * NG_GRP + i) * 2 + 1];
  }
  __syncthreads();
  const float kg2 = kg[g] * kg[g];
  for (int i = tid; i < NG_GRP; i += 256) {
    const float cx = sx[i], cy = sy[i];
    float fx = 0.0f, fy = 0.0f;
    for (int j = 0; j < NG_GRP; ++j) {
      if (j == i) continue;
      const float dx = cx - sx[j], dy = cy - sy[j];
      const float dd = sqrtf(dx * dx + dy * dy) + EPS_F;
      const float coef = kg2 / (dd * dd);
      fx += coef * dx;
      fy += coef * dy;
    }
    const int node = g * NG_GRP + i;
    fnet[node * 2 + 0] += fx;
    fnet[node * 2 + 1] += fy;
  }
}

// ---- per-node: delta = alpha*fnet + ru2(relu(ru1[h||aggR]+b1)); clamp; move -
__global__ void __launch_bounds__(128)
k_node_force(__bf16* __restrict__ Hpad, float* __restrict__ coords,
             const __bf16* __restrict__ AggB,
             const __bf16* __restrict__ W1t /*64x96*/, const float* __restrict__ b1,
             const float* __restrict__ ru2w /*64x2*/, const float* __restrict__ ru2b,
             const float* __restrict__ fnet, const float* __restrict__ alpha_p) {
  __shared__ __attribute__((aligned(16))) __bf16 sW1[64 * 96];
  __shared__ __attribute__((aligned(16))) __bf16 sT[4][16 * 64];
  const int tid = threadIdx.x;
#if defined(HAVE_TDM)
  if (tid == 0) tdm_load_2d(sW1, W1t, 192, 64);
  TDM_WAIT();
#else
  for (int i = tid; i < 64 * 96 / 8; i += 128) ((v8bf*)sW1)[i] = ((const v8bf*)W1t)[i];
#endif
  __syncthreads();
  const int lane = tid & 31, wv = tid >> 5;
  const int hi = lane >> 4, l15 = lane & 15, sub = hi << 3;
  const int nb = (blockIdx.x * 4 + wv) * 16;
  const __bf16* hrow = Hpad + (size_t)(nb + l15) * 32;
  const __bf16* grow = AggB + (size_t)(nb + l15) * 64;
  v8f acc[4] = {};
  for (int k0 = 0; k0 < 96; k0 += 32) {
    v16bf a = (k0 == 0) ? frag_ld(hrow, 0, sub) : frag_ld(grow, k0 - 32, sub);
#pragma unroll
    for (int t = 0; t < 4; ++t) {
      v16bf b = frag_ld(&sW1[(t * 16 + l15) * 96], k0, sub);
      acc[t] = wmma_bf16(a, b, acc[t]);
    }
  }
#pragma unroll
  for (int t = 0; t < 4; ++t) {
    const float bb = b1[t * 16 + l15];
#pragma unroll
    for (int g = 0; g < 8; ++g)
      sT[wv][(g + hi * 8) * 64 + t * 16 + l15] = (__bf16)fmaxf(acc[t][g] + bb, 0.0f);
  }
  if (lane < 16) {
    const int node = nb + lane;
    float dx = ru2b[0], dy = ru2b[1];
    const __bf16* r = &sT[wv][lane * 64];
#pragma unroll
    for (int j = 0; j < 64; ++j) {
      const float v = (float)r[j];
      dx += v * ru2w[j * 2 + 0];
      dy += v * ru2w[j * 2 + 1];
    }
    const float alpha = alpha_p[0];
    dx += alpha * fnet[node * 2 + 0];
    dy += alpha * fnet[node * 2 + 1];
    const float sn = sqrtf(dx * dx + dy * dy);
    const float sc = fminf(CLAMP_F / (sn + 1e-9f), 1.0f);
    const float nx = coords[node * 2 + 0] + dx * sc;
    const float ny = coords[node * 2 + 1] + dy * sc;
    coords[node * 2 + 0] = nx;
    coords[node * 2 + 1] = ny;
    Hpad[(size_t)node * 32 + 0] = (__bf16)nx;
    Hpad[(size_t)node * 32 + 1] = (__bf16)ny;
  }
}

// ---------------------------------------------------------------------------
extern "C" void kernel_launch(void* const* d_in, const int* in_sizes, int n_in,
                              void* d_out, int out_size, void* d_ws, size_t ws_size,
                              hipStream_t stream) {
  (void)in_sizes; (void)out_size; (void)ws_size;
  const float* x    = (const float*)d_in[0];
  const int* ei     = (const int*)d_in[1];
  const int* rowp   = ei;
  const int* colp   = ei + E_TOT;
  const int* batch  = (const int*)d_in[2];
  // params flattened in jax tree order (sorted dict keys, b before w):
  // 3..10: agg0 {m1.b,m1.w,m2.b,m2.w,u1.b,u1.w,u2.b,u2.w}; 11..18: agg1;
  // 19: alpha; 20,21: h1.b,h1.w; 22,23: h2.b,h2.w; 24,25: inp; 26,27: rm1;
  // 28,29: rm2; 30,31: ru1; 32,33: ru2
  const float* prm[34];
  for (int i = 0; i < n_in && i < 34; ++i) prm[i] = (const float*)d_in[i];

  char* ws = (char*)d_ws;
  size_t off = 0;
  auto alloc = [&](size_t b) { size_t o = off; off += (b + 255) & ~(size_t)255; return o; };
  __bf16* W_inp = (__bf16*)(ws + alloc(64 * 32 * 2));
  __bf16 *Wm1[2], *Wm2[2], *Wu1[2], *Wu2[2];
  for (int l = 0; l < 2; ++l) {
    Wm1[l] = (__bf16*)(ws + alloc(64 * 64 * 2));
    Wm2[l] = (__bf16*)(ws + alloc(64 * 64 * 2));
    Wu1[l] = (__bf16*)(ws + alloc(64 * 128 * 2));
    Wu2[l] = (__bf16*)(ws + alloc(64 * 64 * 2));
  }
  __bf16* Wh1  = (__bf16*)(ws + alloc(32 * 128 * 2));
  __bf16* Wrm1 = (__bf16*)(ws + alloc(64 * 32 * 2));
  __bf16* Wrm2 = (__bf16*)(ws + alloc(64 * 64 * 2));
  __bf16* Wru1 = (__bf16*)(ws + alloc(64 * 96 * 2));
  __bf16* Hpad = (__bf16*)(ws + alloc((size_t)N_NODES * 32 * 2));
  float*  crd  = (float*)(ws + alloc((size_t)N_NODES * 2 * 4));
  float*  Zf   = (float*)(ws + alloc((size_t)N_NODES * 64 * 4));
  __bf16* Zb   = (__bf16*)(ws + alloc((size_t)N_NODES * 64 * 2));
  float*  Agg  = (float*)(ws + alloc((size_t)N_NODES * 64 * 4));
  __bf16* AggB = (__bf16*)(ws + alloc((size_t)N_NODES * 64 * 2));
  float*  ewp  = (float*)(ws + alloc((size_t)E_TOT * 4));
  float*  deg  = (float*)(ws + alloc((size_t)N_NODES * 4));
  float*  cnts = (float*)(ws + alloc(G_GRP * 4));
  float*  kgp  = (float*)(ws + alloc(G_GRP * 4));
  float*  fnet = (float*)(ws + alloc((size_t)N_NODES * 2 * 4));

  auto Z32 = [&](float* p, int n) { k_zero_f32<<<(n + 255) / 256, 256, 0, stream>>>(p, n); };
  auto ZB  = [&](__bf16* p, int n) { k_zero_bf<<<(n + 255) / 256, 256, 0, stream>>>(p, n); };
  auto PACK = [&](const float* w, __bf16* o, int Nout, int Kpad, int dk, int sk, int ck) {
    int tot = Nout * ck;
    k_pack_wt<<<(tot + 255) / 256, 256, 0, stream>>>(w, o, Nout, Kpad, dk, sk, ck);
  };

  // weight prep (bf16, transposed, K-padded)
  ZB(W_inp, 64 * 32);  PACK(prm[25], W_inp, 64, 32, 0, 0, 18);
  for (int l = 0; l < 2; ++l) {
    int b = 3 + l * 8;
    PACK(prm[b + 1], Wm1[l], 64, 64, 0, 0, 64);
    PACK(prm[b + 3], Wm2[l], 64, 64, 0, 0, 64);
    PACK(prm[b + 5], Wu1[l], 64, 128, 0, 0, 128);
    PACK(prm[b + 7], Wu2[l], 64, 64, 0, 0, 64);
  }
  PACK(prm[21], Wh1, 32, 128, 0, 0, 128);
  ZB(Wrm1, 64 * 32);  PACK(prm[27], Wrm1, 64, 32, 0, 0, 18);
  PACK(prm[29], Wrm2, 64, 64, 0, 0, 64);
  ZB(Wru1, 64 * 96);
  PACK(prm[31], Wru1, 64, 96, 0, 0, 18);     // h part  -> k 0..17
  PACK(prm[31], Wru1, 64, 96, 32, 18, 64);   // agg part-> k 32..95

  k_pack_h<<<N_NODES / 256, 256, 0, stream>>>(x, Hpad, crd);
  Z32(deg, N_NODES);  Z32(cnts, G_GRP);
  k_deg<<<E_TOT / 256, 256, 0, stream>>>(rowp, deg);
  k_counts<<<N_NODES / 256, 256, 0, stream>>>(batch, cnts);
  k_kg<<<1, 64, 0, stream>>>(cnts, kgp);

  k_node_inp<<<N_NODES / 64, 128, 0, stream>>>(Hpad, W_inp, prm[24], Zf, Zb);

  for (int l = 0; l < 2; ++l) {
    int b = 3 + l * 8;
    Z32(Agg, N_NODES * 64);
    k_edge_mlp<<<E_TOT / 64, 128, 0, stream>>>(Zb, 64, colp, rowp, Wm1[l], prm[b + 0],
                                               Wm2[l], prm[b + 2], Agg);
    k_finish_agg<<<N_NODES * 64 / 256, 256, 0, stream>>>(Agg, deg, AggB);
    k_node_update<<<N_NODES / 64, 128, 0, stream>>>(Zf, Zb, AggB, Wu1[l], prm[b + 4],
                                                    Wu2[l], prm[b + 6]);
  }

  k_head<<<UND_E / 64, 128, 0, stream>>>(Zb, rowp, colp, Wh1, prm[20], prm[21],
                                         prm[23], prm[22], deg, ewp);

  for (int s = 0; s < 3; ++s) {
    Z32(fnet, N_NODES * 2);
    Z32(Agg, N_NODES * 64);
    k_attract<<<E_TOT / 256, 256, 0, stream>>>(crd, rowp, colp, ewp, batch, kgp, fnet);
    k_edge_mlp<<<E_TOT / 64, 128, 0, stream>>>(Hpad, 32, colp, rowp, Wrm1, prm[26],
                                               Wrm2, prm[28], Agg);
    k_finish_agg<<<N_NODES * 64 / 256, 256, 0, stream>>>(Agg, deg, AggB);
    k_repulse<<<G_GRP, 256, 0, stream>>>(crd, kgp, fnet);
    k_node_force<<<N_NODES / 64, 128, 0, stream>>>(Hpad, crd, AggB, Wru1, prm[30],
                                                   prm[33], prm[32], fnet, prm[19]);
  }
  k_copy_out<<<N_NODES * 2 / 256, 256, 0, stream>>>(crd, (float*)d_out);
}
